// StateSequenceTransformer_73452530696211
// MI455X (gfx1250) — compile-verified, matrix operand order checked by reading, not run on hardware
//
#include <hip/hip_runtime.h>
#include <hip/hip_bf16.h>

// ---------------------------------------------------------------------------
// StateSequenceTransformer on MI455X (gfx1250, wave32, WMMA 16x16x32 f16)
//   B=4 T=2048 D_MODEL=128 NHEAD=4 hd=32 DFF=256 WINDOW=64 NLAYERS=3
//   One wave (32 threads) owns a 16-row tile.
//   Weights pre-packed into per-lane B-fragment order -> b128 vector loads.
//   V stored transposed+padded so attn@V B-fragments are contiguous.
// ---------------------------------------------------------------------------

#define R_TOT  8192
#define T_SEQ  2048
#define DM     128
#define DFF_   256
#define QKV_LD 256            /* q (scaled) + k, f16 */
#define T_PAD  2144           /* 64 front pad + 2048 + 32 back pad */
#define LN_EPS 1e-5f
#define QSCALE 0.17677669529663687f /* 1/sqrt(32) */

typedef __attribute__((ext_vector_type(16))) _Float16 v16h;
typedef __attribute__((ext_vector_type(8)))  _Float16 v8h;
typedef __attribute__((ext_vector_type(8)))  float    v8f;

__device__ __forceinline__ v8f wmma16(v16h a, v16h b, v8f c) {
  return __builtin_amdgcn_wmma_f32_16x16x32_f16(false, a, false, b, (short)0, c,
                                                false, false);
}

// A fragment: 16x32 f16 from row-major src (ld multiple of 8, 16B aligned).
// lane<16: row=lane, K={0..7,16..23}; lane>=16: row=lane-16, K={8..15,24..31}
__device__ __forceinline__ v16h load_a_frag(const _Float16* src, int ld) {
  int lane = threadIdx.x & 31;
  const _Float16* p = src + (lane & 15) * ld + ((lane < 16) ? 0 : 8);
  v8h lo = *(const v8h*)p;
  v8h hi = *(const v8h*)(p + 16);
  return __builtin_shufflevector(lo, hi, 0, 1, 2, 3, 4, 5, 6, 7,
                                 8, 9, 10, 11, 12, 13, 14, 15);
}

// Pre-packed B fragment: tile = 512 f16, lane's 16 values contiguous (32B).
__device__ __forceinline__ v16h load_b_packed(const _Float16* base, int tile) {
  int lane = threadIdx.x & 31;
  return *(const v16h*)(base + ((size_t)tile << 9) + (lane << 4));
}

// Build v16h from two contiguous v8h loads at p and p+16 (K^T / generic).
__device__ __forceinline__ v16h load_kpair(const _Float16* p) {
  v8h lo = *(const v8h*)p;
  v8h hi = *(const v8h*)(p + 16);
  return __builtin_shufflevector(lo, hi, 0, 1, 2, 3, 4, 5, 6, 7,
                                 8, 9, 10, 11, 12, 13, 14, 15);
}

// ---------------------------------------------------------------------------
// prep: pack f32 weights (K x N row-major, ld=Nld) into B-fragment tiles.
// tile id = nt*Ktiles + kt; elements outside (Ksrc,Nsrc) -> 0 (padding).
// ---------------------------------------------------------------------------
__global__ void pack_b(const float* __restrict__ src, _Float16* __restrict__ dst,
                       int Ktiles, int Ntiles, int Ksrc, int Nsrc, int Nld) {
  int idx = blockIdx.x * blockDim.x + threadIdx.x;
  int total = Ktiles * Ntiles * 32;
  if (idx >= total) return;
  int lane = idx & 31;
  int tile = idx >> 5;
  int kt = tile % Ktiles, nt = tile / Ktiles;
  int col = lane & 15, kB = (lane < 16) ? 0 : 8;
  int n = nt * 16 + col;
  _Float16* d = dst + ((size_t)tile << 9) + (lane << 4);
#pragma unroll
  for (int i = 0; i < 8; ++i) {
    int k0 = kt * 32 + kB + i, k1 = k0 + 16;
    d[i]     = (k0 < Ksrc && n < Nsrc) ? (_Float16)src[k0 * Nld + n] : (_Float16)0.f;
    d[8 + i] = (k1 < Ksrc && n < Nsrc) ? (_Float16)src[k1 * Nld + n] : (_Float16)0.f;
  }
}

__global__ void build_x16(const float* __restrict__ E, const float* __restrict__ rho,
                          _Float16* __restrict__ x) {
  int i = blockIdx.x * blockDim.x + threadIdx.x;  // R*32
  int row = i >> 5, c = i & 31;
  _Float16 v = (_Float16)0.f;
  if (c < 8)       v = (_Float16)E[row * 8 + c];
  else if (c < 16) v = (_Float16)rho[row * 8 + (c - 8)];
  x[i] = v;
}

__global__ void zero_vT_pads(_Float16* __restrict__ vT) {
  int i = blockIdx.x * blockDim.x + threadIdx.x;  // 512 rows * 96 pad entries
  if (i >= 512 * 96) return;
  int row = i / 96, c = i % 96;
  int pos = (c < 64) ? c : (64 + T_SEQ + (c - 64));
  vT[(size_t)row * T_PAD + pos] = (_Float16)0.f;
}

// ---------------------------------------------------------------------------
// in_proj: h = relu(x@W1+b1)@W2 + b2 + PE  -> h32 + h16
// ---------------------------------------------------------------------------
__global__ void inproj_kernel(const _Float16* __restrict__ x16,
                              const _Float16* __restrict__ pw1,  // 8 tiles
                              const float* __restrict__ b1,
                              const _Float16* __restrict__ pw2,  // 32 tiles [nt*4+kt]
                              const float* __restrict__ b2,
                              float* __restrict__ h32, _Float16* __restrict__ h16) {
  __shared__ _Float16 t1[16 * 128];
  int row0 = blockIdx.x * 16;
  int lane = threadIdx.x & 31;
  int nCol = lane & 15, mBase = (lane < 16) ? 0 : 8;

  v16h a = load_a_frag(x16 + row0 * 32, 32);
#pragma unroll
  for (int nt = 0; nt < 8; ++nt) {
    v8f acc = {};
    acc = wmma16(a, load_b_packed(pw1, nt), acc);
    float bias = b1[nt * 16 + nCol];
#pragma unroll
    for (int r = 0; r < 8; ++r) {
      float v = acc[r] + bias;
      t1[(mBase + r) * 128 + nt * 16 + nCol] = (_Float16)(v > 0.f ? v : 0.f);
    }
  }
  __syncthreads();

  v16h a2[4];
#pragma unroll
  for (int kt = 0; kt < 4; ++kt) a2[kt] = load_a_frag(t1 + kt * 32, 128);
#pragma unroll
  for (int nt = 0; nt < 8; ++nt) {
    v8f acc = {};
#pragma unroll
    for (int kt = 0; kt < 4; ++kt)
      acc = wmma16(a2[kt], load_b_packed(pw2, nt * 4 + kt), acc);
    int n = nt * 16 + nCol;
    float bias = b2[n];
    float div = __expf(-(float)(n >> 1) * (9.210340371976184f / 64.f));
#pragma unroll
    for (int r = 0; r < 8; ++r) {
      int row = row0 + mBase + r;
      float ang = (float)(row & (T_SEQ - 1)) * div;
      float pe = (n & 1) ? __cosf(ang) : __sinf(ang);
      float v = acc[r] + bias + pe;
      h32[row * DM + n] = v;
      h16[row * DM + n] = (_Float16)v;
    }
  }
}

// ---------------------------------------------------------------------------
// qkv: q (scaled) and k -> qkv16 (R x 256); v -> vT transposed+padded.
// grid (R/16, 24)
// ---------------------------------------------------------------------------
__global__ void qkv_kernel(const _Float16* __restrict__ h16,
                           const _Float16* __restrict__ pqkv,  // 96 tiles [nt*4+kt]
                           const float* __restrict__ bqkv,
                           _Float16* __restrict__ qkv16, _Float16* __restrict__ vT) {
  int row0 = blockIdx.x * 16;
  int nt = blockIdx.y;
  int lane = threadIdx.x & 31;
  int nCol = lane & 15, mBase = (lane < 16) ? 0 : 8;

  v8f acc = {};
#pragma unroll
  for (int kt = 0; kt < 4; ++kt)
    acc = wmma16(load_a_frag(h16 + row0 * DM + kt * 32, DM),
                 load_b_packed(pqkv, nt * 4 + kt), acc);
  int n = nt * 16 + nCol;
  float bias = bqkv[n];
  if (n < 256) {  // q (scaled) / k : row-major
    float scale = (n < 128) ? QSCALE : 1.0f;
#pragma unroll
    for (int r = 0; r < 8; ++r)
      qkv16[(size_t)(row0 + mBase + r) * QKV_LD + n] =
          (_Float16)((acc[r] + bias) * scale);
  } else {        // v : transposed with padded time axis, vectorized store
    int d = n - 256;
    int head = d >> 5, dd = d & 31;
    int bIdx = row0 >> 11;
    int t0m = (row0 & (T_SEQ - 1)) + mBase;
    v8h vv;
#pragma unroll
    for (int r = 0; r < 8; ++r) vv[r] = (_Float16)(acc[r] + bias);
    *(v8h*)(vT + (size_t)((bIdx * 4 + head) * 32 + dd) * T_PAD + 64 + t0m) = vv;
  }
}

// ---------------------------------------------------------------------------
// windowed attention: 16 queries x 1 head per wave. grid (R/16, NHEAD)
// ---------------------------------------------------------------------------
__global__ void attn_kernel(const _Float16* __restrict__ qkv16,
                            const _Float16* __restrict__ vT,
                            _Float16* __restrict__ ao16) {
  __shared__ _Float16 at[16 * 96];  // attn probs, K padded 80 -> 96
  int row0 = blockIdx.x * 16;
  int head = blockIdx.y;
  int t0 = row0 & (T_SEQ - 1);
  int bRow0 = row0 - t0;
  int bIdx = row0 >> 11;
  int lane = threadIdx.x & 31;
  int nCol = lane & 15, mBase = (lane < 16) ? 0 : 8, kBase = mBase;
  int keyBase = t0 - 64;

  v16h qa = load_a_frag(qkv16 + (size_t)row0 * QKV_LD + head * 32, QKV_LD);

  v8f s[5];
#pragma unroll
  for (int kt = 0; kt < 5; ++kt) {
    int j = keyBase + kt * 16 + nCol;   // key position for this lane's column
    int jr = j < 0 ? 0 : j;             // clamp address; mask handles validity
    v16h kb = load_kpair(qkv16 + (size_t)(bRow0 + jr) * QKV_LD + 128 + head * 32 + kBase);
    v8f c = {};
    c = wmma16(qa, kb, c);
#pragma unroll
    for (int r = 0; r < 8; ++r) {
      int iQ = t0 + mBase + r;
      bool valid = (j >= 0) && (j <= iQ) && (j >= iQ - 63);
      s[kt][r] = valid ? c[r] : -1e30f;
    }
  }

  // row softmax: row m lives in one vgpr slot across the 16 lanes of a half
#pragma unroll
  for (int r = 0; r < 8; ++r) {
    float mx = s[0][r];
#pragma unroll
    for (int kt = 1; kt < 5; ++kt) mx = fmaxf(mx, s[kt][r]);
#pragma unroll
    for (int msk = 1; msk < 16; msk <<= 1) mx = fmaxf(mx, __shfl_xor(mx, msk, 32));
    float sum = 0.f;
#pragma unroll
    for (int kt = 0; kt < 5; ++kt) { float e = __expf(s[kt][r] - mx); s[kt][r] = e; sum += e; }
#pragma unroll
    for (int msk = 1; msk < 16; msk <<= 1) sum += __shfl_xor(sum, msk, 32);
    float inv = 1.0f / sum;
#pragma unroll
    for (int kt = 0; kt < 5; ++kt) s[kt][r] *= inv;
  }

  // C-layout -> A-fragment layout via LDS (16 x 96, cols 80..95 zero)
#pragma unroll
  for (int kt = 0; kt < 5; ++kt)
#pragma unroll
    for (int r = 0; r < 8; ++r)
      at[(mBase + r) * 96 + kt * 16 + nCol] = (_Float16)s[kt][r];
#pragma unroll
  for (int r = 0; r < 8; ++r) at[(mBase + r) * 96 + 80 + nCol] = (_Float16)0.f;
  __syncthreads();

  // attn(16x96) @ V(96x32): 2 N-tiles x 3 K-chunks, contiguous vT loads
#pragma unroll
  for (int nt = 0; nt < 2; ++nt) {
    int d = nt * 16 + nCol;
    const _Float16* vrow = vT + (size_t)((bIdx * 4 + head) * 32 + d) * T_PAD + 64 + keyBase;
    v8f acc = {};
#pragma unroll
    for (int kc = 0; kc < 3; ++kc) {
      v16h aa = load_a_frag(at + kc * 32, 96);
      v16h vb = load_kpair(vrow + kc * 32 + kBase);
      acc = wmma16(aa, vb, acc);
    }
#pragma unroll
    for (int r = 0; r < 8; ++r)
      ao16[(size_t)(row0 + mBase + r) * DM + head * 32 + nt * 16 + nCol] = (_Float16)acc[r];
  }
}

// ---------------------------------------------------------------------------
// o-proj + residual + layernorm -> h32/h16
// ---------------------------------------------------------------------------
__global__ void oproj_ln_kernel(const _Float16* __restrict__ ao16,
                                const _Float16* __restrict__ pwo,  // 32 tiles
                                const float* __restrict__ bo,
                                const float* __restrict__ g, const float* __restrict__ bb,
                                float* __restrict__ h32, _Float16* __restrict__ h16) {
  int row0 = blockIdx.x * 16;
  int lane = threadIdx.x & 31;
  int nCol = lane & 15, mBase = (lane < 16) ? 0 : 8;

  v16h a[4];
#pragma unroll
  for (int kt = 0; kt < 4; ++kt) a[kt] = load_a_frag(ao16 + row0 * DM + kt * 32, DM);

  v8f out[8];
#pragma unroll
  for (int nt = 0; nt < 8; ++nt) {
    v8f acc = {};
#pragma unroll
    for (int kt = 0; kt < 4; ++kt)
      acc = wmma16(a[kt], load_b_packed(pwo, nt * 4 + kt), acc);
    float bias = bo[nt * 16 + nCol];
#pragma unroll
    for (int r = 0; r < 8; ++r)
      out[nt][r] = acc[r] + bias + h32[(row0 + mBase + r) * DM + nt * 16 + nCol];
  }
#pragma unroll
  for (int r = 0; r < 8; ++r) {
    float sm = 0.f, sq = 0.f;
#pragma unroll
    for (int nt = 0; nt < 8; ++nt) { float v = out[nt][r]; sm += v; sq += v * v; }
#pragma unroll
    for (int msk = 1; msk < 16; msk <<= 1) { sm += __shfl_xor(sm, msk, 32); sq += __shfl_xor(sq, msk, 32); }
    float mu = sm * (1.f / 128.f);
    float inv = rsqrtf(sq * (1.f / 128.f) - mu * mu + LN_EPS);
#pragma unroll
    for (int nt = 0; nt < 8; ++nt) {
      int n = nt * 16 + nCol;
      int row = row0 + mBase + r;
      float y = (out[nt][r] - mu) * inv * g[n] + bb[n];
      h32[row * DM + n] = y;
      h16[row * DM + n] = (_Float16)y;
    }
  }
}

// ---------------------------------------------------------------------------
// FFN: relu(h@W1+b1)@W2+b2, residual + layernorm
// ---------------------------------------------------------------------------
__global__ void ffn_ln_kernel(const _Float16* __restrict__ h16in,
                              const _Float16* __restrict__ pf1,  // 64 tiles [nt*4+kt]
                              const float* __restrict__ b1,
                              const _Float16* __restrict__ pf2,  // 64 tiles [nt*8+kt]
                              const float* __restrict__ b2,
                              const float* __restrict__ g, const float* __restrict__ bb,
                              float* __restrict__ h32, _Float16* __restrict__ h16) {
  __shared__ _Float16 mid[16 * DFF_];
  int row0 = blockIdx.x * 16;
  int lane = threadIdx.x & 31;
  int nCol = lane & 15, mBase = (lane < 16) ? 0 : 8;

  v16h a[4];
#pragma unroll
  for (int kt = 0; kt < 4; ++kt) a[kt] = load_a_frag(h16in + row0 * DM + kt * 32, DM);
#pragma unroll
  for (int nt = 0; nt < 16; ++nt) {
    v8f acc = {};
#pragma unroll
    for (int kt = 0; kt < 4; ++kt)
      acc = wmma16(a[kt], load_b_packed(pf1, nt * 4 + kt), acc);
    float bias = b1[nt * 16 + nCol];
#pragma unroll
    for (int r = 0; r < 8; ++r) {
      float v = acc[r] + bias;
      mid[(mBase + r) * DFF_ + nt * 16 + nCol] = (_Float16)(v > 0.f ? v : 0.f);
    }
  }
  __syncthreads();

  v16h a2[8];
#pragma unroll
  for (int kt = 0; kt < 8; ++kt) a2[kt] = load_a_frag(mid + kt * 32, DFF_);

  v8f out[8];
#pragma unroll
  for (int nt = 0; nt < 8; ++nt) {
    v8f acc = {};
#pragma unroll
    for (int kt = 0; kt < 8; ++kt)
      acc = wmma16(a2[kt], load_b_packed(pf2, nt * 8 + kt), acc);
    float bias = b2[nt * 16 + nCol];
#pragma unroll
    for (int r = 0; r < 8; ++r)
      out[nt][r] = acc[r] + bias + h32[(row0 + mBase + r) * DM + nt * 16 + nCol];
  }
#pragma unroll
  for (int r = 0; r < 8; ++r) {
    float sm = 0.f, sq = 0.f;
#pragma unroll
    for (int nt = 0; nt < 8; ++nt) { float v = out[nt][r]; sm += v; sq += v * v; }
#pragma unroll
    for (int msk = 1; msk < 16; msk <<= 1) { sm += __shfl_xor(sm, msk, 32); sq += __shfl_xor(sq, msk, 32); }
    float mu = sm * (1.f / 128.f);
    float inv = rsqrtf(sq * (1.f / 128.f) - mu * mu + LN_EPS);
#pragma unroll
    for (int nt = 0; nt < 8; ++nt) {
      int n = nt * 16 + nCol;
      int row = row0 + mBase + r;
      float y = (out[nt][r] - mu) * inv * g[n] + bb[n];
      h32[row * DM + n] = y;
      h16[row * DM + n] = (_Float16)y;
    }
  }
}

// ---------------------------------------------------------------------------
// head: out = h16 @ W_head(128x16 padded) + b_head -> fp32 (R x 8)
// ---------------------------------------------------------------------------
__global__ void head_kernel(const _Float16* __restrict__ h16,
                            const _Float16* __restrict__ pwh,  // 4 tiles
                            const float* __restrict__ bh,
                            float* __restrict__ out) {
  int row0 = blockIdx.x * 16;
  int lane = threadIdx.x & 31;
  int nCol = lane & 15, mBase = (lane < 16) ? 0 : 8;
  v8f acc = {};
#pragma unroll
  for (int kt = 0; kt < 4; ++kt)
    acc = wmma16(load_a_frag(h16 + row0 * DM + kt * 32, DM),
                 load_b_packed(pwh, kt), acc);
  if (nCol < 8) {
    float bias = bh[nCol];
#pragma unroll
    for (int r = 0; r < 8; ++r) out[(row0 + mBase + r) * 8 + nCol] = acc[r] + bias;
  }
}

// ---------------------------------------------------------------------------
// host launch
// ---------------------------------------------------------------------------
extern "C" void kernel_launch(void* const* d_in, const int* in_sizes, int n_in,
                              void* d_out, int out_size, void* d_ws, size_t ws_size,
                              hipStream_t stream) {
  (void)in_sizes; (void)n_in; (void)out_size; (void)ws_size;
  const float* E      = (const float*)d_in[0];
  const float* rho    = (const float*)d_in[1];
  const float* W_in1  = (const float*)d_in[2];
  const float* b_in1  = (const float*)d_in[3];
  const float* W_in2  = (const float*)d_in[4];
  const float* b_in2  = (const float*)d_in[5];
  const float* Wqkv   = (const float*)d_in[6];
  const float* bqkv   = (const float*)d_in[7];
  const float* Wo     = (const float*)d_in[8];
  const float* bo     = (const float*)d_in[9];
  const float* ln1_g  = (const float*)d_in[10];
  const float* ln1_b  = (const float*)d_in[11];
  const float* Wff1   = (const float*)d_in[12];
  const float* bff1   = (const float*)d_in[13];
  const float* Wff2   = (const float*)d_in[14];
  const float* bff2   = (const float*)d_in[15];
  const float* ln2_g  = (const float*)d_in[16];
  const float* ln2_b  = (const float*)d_in[17];
  const float* W_head = (const float*)d_in[18];
  const float* b_head = (const float*)d_in[19];

  char* w = (char*)d_ws;
  auto take = [&](size_t bytes) -> void* {
    void* p = (void*)w;
    w += (bytes + 255) & ~(size_t)255;
    return p;
  };
  _Float16* x16   = (_Float16*)take((size_t)R_TOT * 32 * 2);
  _Float16* h16   = (_Float16*)take((size_t)R_TOT * DM * 2);
  float*    h32   = (float*)   take((size_t)R_TOT * DM * 4);
  _Float16* qkv16 = (_Float16*)take((size_t)R_TOT * QKV_LD * 2);
  _Float16* vT    = (_Float16*)take((size_t)512 * T_PAD * 2);  // 16 (b*head) * 32 d rows
  _Float16* ao16  = (_Float16*)take((size_t)R_TOT * DM * 2);
  _Float16* pw1   = (_Float16*)take((size_t)8 * 512 * 2);
  _Float16* pw2   = (_Float16*)take((size_t)32 * 512 * 2);
  _Float16* pqkv  = (_Float16*)take((size_t)3 * 96 * 512 * 2);
  _Float16* pwo   = (_Float16*)take((size_t)3 * 32 * 512 * 2);
  _Float16* pf1   = (_Float16*)take((size_t)3 * 64 * 512 * 2);
  _Float16* pf2   = (_Float16*)take((size_t)3 * 64 * 512 * 2);
  _Float16* pwh   = (_Float16*)take((size_t)4 * 512 * 2);

  auto cdiv = [](int a, int b) { return (a + b - 1) / b; };

  // weight packing (f32 -> f16 B-fragment tiles, with K/N zero-padding)
  pack_b<<<cdiv(1 * 8 * 32, 256), 256, 0, stream>>>(W_in1, pw1, 1, 8, 16, 128, 128);
  pack_b<<<cdiv(4 * 8 * 32, 256), 256, 0, stream>>>(W_in2, pw2, 4, 8, 128, 128, 128);
  for (int l = 0; l < 3; ++l) {
    pack_b<<<cdiv(4 * 24 * 32, 256), 256, 0, stream>>>(
        Wqkv + (size_t)l * 128 * 384, pqkv + (size_t)l * 96 * 512, 4, 24, 128, 384, 384);
    pack_b<<<cdiv(4 * 8 * 32, 256), 256, 0, stream>>>(
        Wo + (size_t)l * 128 * 128, pwo + (size_t)l * 32 * 512, 4, 8, 128, 128, 128);
    pack_b<<<cdiv(4 * 16 * 32, 256), 256, 0, stream>>>(
        Wff1 + (size_t)l * 128 * 256, pf1 + (size_t)l * 64 * 512, 4, 16, 128, 256, 256);
    pack_b<<<cdiv(8 * 8 * 32, 256), 256, 0, stream>>>(
        Wff2 + (size_t)l * 256 * 128, pf2 + (size_t)l * 64 * 512, 8, 8, 256, 128, 128);
  }
  pack_b<<<cdiv(4 * 1 * 32, 256), 256, 0, stream>>>(W_head, pwh, 4, 1, 128, 8, 8);

  build_x16<<<R_TOT * 32 / 256, 256, 0, stream>>>(E, rho, x16);
  zero_vT_pads<<<cdiv(512 * 96, 256), 256, 0, stream>>>(vT);

  const int TILES = R_TOT / 16;  // 512
  inproj_kernel<<<TILES, 32, 0, stream>>>(x16, pw1, b_in1, pw2, b_in2, h32, h16);

  for (int l = 0; l < 3; ++l) {
    qkv_kernel<<<dim3(TILES, 24), 32, 0, stream>>>(
        h16, pqkv + (size_t)l * 96 * 512, bqkv + l * 384, qkv16, vT);
    attn_kernel<<<dim3(TILES, 4), 32, 0, stream>>>(qkv16, vT, ao16);
    oproj_ln_kernel<<<TILES, 32, 0, stream>>>(
        ao16, pwo + (size_t)l * 32 * 512, bo + l * 128,
        ln1_g + l * 128, ln1_b + l * 128, h32, h16);
    ffn_ln_kernel<<<TILES, 32, 0, stream>>>(
        h16, pf1 + (size_t)l * 64 * 512, bff1 + l * 256,
        pf2 + (size_t)l * 64 * 512, bff2 + l * 128,
        ln2_g + l * 128, ln2_b + l * 128, h32, h16);
  }
  head_kernel<<<TILES, 32, 0, stream>>>(h16, pwh, b_head, (float*)d_out);
}
